// CrossAttentionLayer_10857677324874
// MI455X (gfx1250) — matrix operand order repaired and checked by reading, not compiled, not run on hardware
//
#include <hip/hip_runtime.h>
#include <hip/hip_bf16.h>

typedef __bf16 bf16_t;
typedef __attribute__((ext_vector_type(16))) __bf16 v16bf;
typedef __attribute__((ext_vector_type(8)))  __bf16 v8bf;
typedef __attribute__((ext_vector_type(8)))  float  v8f;

#define NB    8
#define LSEQ  4096
#define CH    512
#define CH2   1024
#define NHEAD 8
#define HDIM  64
#define KVS   64
#define TILE  64
#define SLS   40   // padded halves per column in slab-major weight layout

// slab sizes (elements): one k-step slab of 512 columns
#define SLAB_ELEMS (512 * SLS)          // 20480 halves = 40 KB
#define SLAB_CHUNKS (SLAB_ELEMS / 8)    // 2560 v8bf chunks
#define CHUNKS_PER_T (SLAB_CHUNKS / 256)// 10 per thread

// ---------------- workspace layout (bytes) ----------------
static constexpr size_t O_GFEAT = 0;                                   // 8*512*4
static constexpr size_t O_H     = O_GFEAT + (size_t)NB*CH*4;           // 8*1024*4
static constexpr size_t O_MEM   = O_H     + (size_t)NB*CH2*4;          // 8*128*512*4
static constexpr size_t O_KMAT  = O_MEM   + (size_t)NB*2*KVS*CH*4;     // 8*64*512*4
static constexpr size_t O_VMAT  = O_KMAT  + (size_t)NB*KVS*CH*4;
static constexpr size_t O_KSUM  = O_VMAT  + (size_t)NB*KVS*CH*4;       // 8*8*64*4
static constexpr size_t O_KVT   = O_KSUM  + (size_t)NB*NHEAD*HDIM*4;   // bf16 [n][h][v][d]
// slab-major bf16 weights: [K/32][N][SLS]
static constexpr size_t O_QW    = O_KVT + (size_t)NB*NHEAD*HDIM*HDIM*2;
static constexpr size_t O_MW    = O_QW  + (size_t)(CH/32)*CH*SLS*2;    // 655360
static constexpr size_t O_W1    = O_MW  + (size_t)(CH/32)*CH*SLS*2;
static constexpr size_t O_W2    = O_W1  + (size_t)(CH2/32)*CH2*SLS*2;  // 2621440
static constexpr size_t O_X16   = O_W2  + (size_t)(CH2/32)*CH*SLS*2;   // 1310720
static constexpr size_t O_MSG   = O_X16 + (size_t)NB*LSEQ*CH*2;

// ---------------- helpers ----------------
__device__ __forceinline__ v8f zero8() { v8f z = {}; return z; }

__device__ __forceinline__ float elu1(float x) { return x > 0.f ? x + 1.f : __expf(x); }

__device__ __forceinline__ v16bf mk16(v8bf lo, v8bf hi) {
  v16bf r;
#pragma unroll
  for (int e = 0; e < 8; ++e) { r[e] = lo[e]; r[e + 8] = hi[e]; }
  return r;
}

// A fragment: 16x32 bf16, row-major source, lane=M in 0..15, hi half-wave => K+16
__device__ __forceinline__ v16bf load_a_frag(const bf16_t* base, int lda, int lane) {
  const bf16_t* row = base + (size_t)(lane & 15) * lda + ((lane >> 4) << 4);
  return mk16(*(const v8bf*)row, *(const v8bf*)(row + 8));
}

// B fragment from slab (column n contiguous in k, padded stride SLS)
__device__ __forceinline__ v16bf load_b_slab(const bf16_t* slab, int ncol, int lane) {
  const bf16_t* p = slab + (size_t)(ncol + (lane & 15)) * SLS + ((lane >> 4) << 4);
  return mk16(*(const v8bf*)p, *(const v8bf*)(p + 8));
}

__device__ __forceinline__ v8f wmma_bf(v16bf a, v16bf b, v8f c) {
  return __builtin_amdgcn_wmma_f32_16x16x32_bf16(false, a, false, b, (short)0, c,
                                                 false, false);
}

// one k-step: 4 Mtiles x 4 Ntiles for this wave's 64-col chunk
__device__ __forceinline__ void gemm_tiles(const bf16_t* a_base, int lda,
                                           const bf16_t* slab, int wv, int lane,
                                           v8f acc[4][4]) {
  v16bf a[4];
#pragma unroll
  for (int mt = 0; mt < 4; ++mt)
    a[mt] = load_a_frag(a_base + (size_t)mt * 16 * lda, lda, lane);
#pragma unroll
  for (int nt = 0; nt < 4; ++nt) {
    v16bf b = load_b_slab(slab, wv * 64 + nt * 16, lane);
#pragma unroll
    for (int mt = 0; mt < 4; ++mt) acc[mt][nt] = wmma_bf(a[mt], b, acc[mt][nt]);
  }
}

__device__ __forceinline__ void slab_gload(v8bf pref[CHUNKS_PER_T],
                                           const bf16_t* wsrc, int t) {
#pragma unroll
  for (int j = 0; j < CHUNKS_PER_T; ++j)
    pref[j] = *(const v8bf*)(wsrc + ((size_t)(t + j * 256) << 3));
}

__device__ __forceinline__ void slab_store(bf16_t* slab, const v8bf pref[CHUNKS_PER_T],
                                           int t) {
#pragma unroll
  for (int j = 0; j < CHUNKS_PER_T; ++j)
    *(v8bf*)(slab + ((size_t)(t + j * 256) << 3)) = pref[j];
}

// double-buffered pipelined GEMM over nk (even) k-steps of 32.
// Ping/pong buffers are referenced explicitly so LDS address-space inference
// succeeds and slab traffic lowers to ds_load/ds_store (not flat).
__device__ __forceinline__ void gemm_db(const bf16_t* a_base, int lda, int nk,
                                        const bf16_t* wbase, size_t slab_pitch,
                                        bf16_t* slab0, bf16_t* slab1,
                                        int t, int wv, int lane, v8f acc[4][4]) {
  v8bf pref[CHUNKS_PER_T];
  slab_gload(pref, wbase, t);
  slab_store(slab0, pref, t);
  for (int s = 0; s < nk; s += 2) {
    __syncthreads();
    if (s + 1 < nk) slab_gload(pref, wbase + (size_t)(s + 1) * slab_pitch, t);
    gemm_tiles(a_base + (size_t)s * 32, lda, slab0, wv, lane, acc);
    if (s + 1 < nk) slab_store(slab1, pref, t);
    __syncthreads();
    if (s + 2 < nk) slab_gload(pref, wbase + (size_t)(s + 2) * slab_pitch, t);
    gemm_tiles(a_base + (size_t)(s + 1) * 32, lda, slab1, wv, lane, acc);
    if (s + 2 < nk) slab_store(slab0, pref, t);
  }
  __syncthreads();
}

// ---------------- small scalar kernels ----------------
__global__ __launch_bounds__(512) void mean_kernel(const float* __restrict__ x,
                                                   float* __restrict__ gfeat) {
  int n = blockIdx.x, c = threadIdx.x;
  const float* p = x + (size_t)n * LSEQ * CH + c;
  float s = 0.f;
  for (int l = 0; l < LSEQ; ++l) s += p[(size_t)l * CH];
  gfeat[n * CH + c] = s * (1.f / LSEQ);
}

__global__ __launch_bounds__(1024) void hidden_kernel(const float* __restrict__ gfeat,
                                                      const float* __restrict__ w1,
                                                      const float* __restrict__ b1,
                                                      float* __restrict__ h) {
  int n = blockIdx.x, j = threadIdx.x;
  float s = b1[j];
  for (int k = 0; k < CH; ++k) s += gfeat[n * CH + k] * w1[(size_t)k * CH2 + j];
  h[n * CH2 + j] = fmaxf(s, 0.f);
}

__global__ __launch_bounds__(256) void memgen_kernel(const float* __restrict__ h,
                                                     const float* __restrict__ w2,
                                                     const float* __restrict__ b2,
                                                     float* __restrict__ mem) {
  int j = blockIdx.x * 256 + threadIdx.x;  // 0..65535
  float acc[NB];
#pragma unroll
  for (int n = 0; n < NB; ++n) acc[n] = 0.f;
  for (int k = 0; k < CH2; ++k) {
    float w = w2[(size_t)k * (CH * KVS * 2) + j];
#pragma unroll
    for (int n = 0; n < NB; ++n) acc[n] += h[n * CH2 + k] * w;
  }
  float b = b2[j];
#pragma unroll
  for (int n = 0; n < NB; ++n) mem[(size_t)n * (CH * KVS * 2) + j] = acc[n] + b;
}

__global__ __launch_bounds__(256) void kvproj_kernel(const float* __restrict__ mem,
                                                     const float* __restrict__ kw,
                                                     const float* __restrict__ vw,
                                                     float* __restrict__ Kmat,
                                                     float* __restrict__ Vmat) {
  int bid = blockIdx.x;  // n*64 + s
  int n = bid >> 6, s = bid & 63, t = threadIdx.x;
  const float* krow = mem + (size_t)n * (CH * KVS * 2) + (size_t)s * CH;
  const float* vrow = mem + (size_t)n * (CH * KVS * 2) + (size_t)(KVS + s) * CH;
  int c0 = t, c1 = t + 256;
  float ka0 = 0.f, ka1 = 0.f, va0 = 0.f, va1 = 0.f;
  for (int k = 0; k < CH; ++k) {
    float kk = krow[k], vv = vrow[k];
    const float* kwr = kw + (size_t)k * CH;
    const float* vwr = vw + (size_t)k * CH;
    ka0 += kk * kwr[c0]; ka1 += kk * kwr[c1];
    va0 += vv * vwr[c0]; va1 += vv * vwr[c1];
  }
  size_t base = (size_t)(n * KVS + s) * CH;
  Kmat[base + c0] = elu1(ka0);  Kmat[base + c1] = elu1(ka1);
  Vmat[base + c0] = va0;        Vmat[base + c1] = va1;
}

// per (n,h): Ksum[d] = sum_s K; KVt[v][d] = sum_s K[s,d]*V[s,v] (bf16, B-ready)
__global__ __launch_bounds__(256) void kvred_kernel(const float* __restrict__ Kmat,
                                                    const float* __restrict__ Vmat,
                                                    float* __restrict__ ksumO,
                                                    bf16_t* __restrict__ kvt) {
  __shared__ float kbuf[HDIM], vbuf[HDIM];
  int n = blockIdx.x >> 3, hh = blockIdx.x & 7, t = threadIdx.x;
  float acc[16];
#pragma unroll
  for (int i = 0; i < 16; ++i) acc[i] = 0.f;
  float ks = 0.f;
  for (int s = 0; s < KVS; ++s) {
    __syncthreads();
    if (t < HDIM)          kbuf[t] = Kmat[(size_t)(n * KVS + s) * CH + hh * HDIM + t];
    else if (t < 2 * HDIM) vbuf[t - HDIM] = Vmat[(size_t)(n * KVS + s) * CH + hh * HDIM + (t - HDIM)];
    __syncthreads();
#pragma unroll
    for (int i = 0; i < 16; ++i) {
      int idx = t + (i << 8);
      acc[i] += kbuf[idx & 63] * vbuf[idx >> 6];
    }
    if (t < HDIM) ks += kbuf[t];
  }
#pragma unroll
  for (int i = 0; i < 16; ++i) {
    int idx = t + (i << 8);
    int d = idx & 63, v = idx >> 6;
    kvt[((size_t)(n * NHEAD + hh) * HDIM + v) * HDIM + d] = (bf16_t)acc[i];
  }
  if (t < HDIM) ksumO[(n * NHEAD + hh) * HDIM + t] = ks;
}

// plain f32 -> bf16
__global__ __launch_bounds__(256) void cvt_kernel(const float* __restrict__ src,
                                                  bf16_t* __restrict__ dst, size_t nel) {
  size_t i = (size_t)blockIdx.x * 256 + threadIdx.x;
  if (i < nel) dst[i] = (bf16_t)src[i];
}

// f32 row-major [K][N] -> slab-major transposed bf16 [K/32][N][SLS]
__global__ __launch_bounds__(256) void cvtw_kernel(const float* __restrict__ src,
                                                   bf16_t* __restrict__ dst,
                                                   int K, int N) {
  size_t i = (size_t)blockIdx.x * 256 + threadIdx.x;
  if (i >= (size_t)K * N) return;
  int k = (int)(i / N);
  int n = (int)(i - (size_t)k * N);
  dst[((size_t)(k >> 5) * N + n) * SLS + (k & 31)] = (bf16_t)src[i];
}

// ---------------- fused q-proj + linear attention + merge + LN1 ----------------
__global__ __launch_bounds__(256) void qattn_kernel(
    const bf16_t* __restrict__ x16, const bf16_t* __restrict__ qw,
    const float* __restrict__ ksum, const bf16_t* __restrict__ kvt,
    const bf16_t* __restrict__ mw, const float* __restrict__ g1,
    const float* __restrict__ b1, bf16_t* __restrict__ msg16) {
  __shared__ bf16_t xls[TILE * CH];
  __shared__ bf16_t qls[TILE * CH];
  __shared__ bf16_t slabA[SLAB_ELEMS];
  __shared__ bf16_t slabB[SLAB_ELEMS];
  __shared__ float  zbuf[TILE * NHEAD];
  __shared__ float  rsum[TILE], rsq[TILE];

  int t = threadIdx.x, lane = t & 31, wv = t >> 5;
  int n = blockIdx.x >> 6, tb = blockIdx.x & 63;
  int hi8 = (lane >> 4) << 3, cl = lane & 15;
  const bf16_t* xsrc = x16 + (size_t)(n * LSEQ + tb * TILE) * CH;

  for (int i = t; i < TILE * CH / 8; i += 256)
    ((v8bf*)xls)[i] = ((const v8bf*)xsrc)[i];
  if (t < TILE) { rsum[t] = 0.f; rsq[t] = 0.f; }
  __syncthreads();

  // GEMM1: q = x @ q_w  -> elu+1 -> qls
  v8f acc[4][4];
#pragma unroll
  for (int mt = 0; mt < 4; ++mt)
#pragma unroll
    for (int nt = 0; nt < 4; ++nt) acc[mt][nt] = zero8();
  gemm_db(xls, CH, CH / 32, qw, (size_t)CH * SLS, slabA, slabB, t, wv, lane, acc);

#pragma unroll
  for (int mt = 0; mt < 4; ++mt)
#pragma unroll
    for (int nt = 0; nt < 4; ++nt)
#pragma unroll
      for (int r = 0; r < 8; ++r) {
        int m = mt * 16 + r + hi8;
        int c = wv * 64 + nt * 16 + cl;
        qls[m * CH + c] = (bf16_t)elu1(acc[mt][nt][r]);
      }
  __syncthreads();

  // Z = 1 / (Q . Ksum + eps)
  for (int i = t; i < TILE * NHEAD; i += 256) {
    int l = i >> 3, hh = i & 7;
    const float* kp = ksum + (n * NHEAD + hh) * HDIM;
    float dot = 0.f;
#pragma unroll 8
    for (int d = 0; d < HDIM; ++d) dot += (float)qls[l * CH + hh * HDIM + d] * kp[d];
    zbuf[i] = 1.f / (dot + 1e-6f);
  }
  __syncthreads();

  // attention apply: wave wv handles head h=wv: out = (Q_h @ KV_h) * Z -> xls
  {
    int hh = wv;
    const bf16_t* kvbase = kvt + (size_t)(n * NHEAD + hh) * HDIM * HDIM;
    v8f a2[4][4];
#pragma unroll
    for (int mt = 0; mt < 4; ++mt)
#pragma unroll
      for (int nt = 0; nt < 4; ++nt) a2[mt][nt] = zero8();
#pragma unroll
    for (int k0 = 0; k0 < HDIM; k0 += 32) {
      v16bf a[4];
#pragma unroll
      for (int mt = 0; mt < 4; ++mt)
        a[mt] = load_a_frag(qls + (size_t)mt * 16 * CH + hh * HDIM + k0, CH, lane);
#pragma unroll
      for (int nt = 0; nt < 4; ++nt) {
        const bf16_t* colb = kvbase + (size_t)(nt * 16 + cl) * HDIM + k0 + ((lane >> 4) << 4);
        v16bf b = mk16(*(const v8bf*)colb, *(const v8bf*)(colb + 8));
#pragma unroll
        for (int mt = 0; mt < 4; ++mt) a2[mt][nt] = wmma_bf(a[mt], b, a2[mt][nt]);
      }
    }
#pragma unroll
    for (int mt = 0; mt < 4; ++mt)
#pragma unroll
      for (int nt = 0; nt < 4; ++nt)
#pragma unroll
        for (int r = 0; r < 8; ++r) {
          int m = mt * 16 + r + hi8;
          xls[m * CH + hh * HDIM + nt * 16 + cl] =
              (bf16_t)(a2[mt][nt][r] * zbuf[m * NHEAD + hh]);
        }
  }
  __syncthreads();

  // GEMM2: msg = attn_out @ merge_w -> layernorm(norm1) -> bf16 msg
#pragma unroll
  for (int mt = 0; mt < 4; ++mt)
#pragma unroll
    for (int nt = 0; nt < 4; ++nt) acc[mt][nt] = zero8();
  gemm_db(xls, CH, CH / 32, mw, (size_t)CH * SLS, slabA, slabB, t, wv, lane, acc);

#pragma unroll
  for (int mt = 0; mt < 4; ++mt)
#pragma unroll
    for (int r = 0; r < 8; ++r) {
      int m = mt * 16 + r + hi8;
      float s1 = 0.f, s2 = 0.f;
#pragma unroll
      for (int nt = 0; nt < 4; ++nt) { float v = acc[mt][nt][r]; s1 += v; s2 += v * v; }
      atomicAdd(&rsum[m], s1);
      atomicAdd(&rsq[m], s2);
    }
  __syncthreads();
#pragma unroll
  for (int mt = 0; mt < 4; ++mt)
#pragma unroll
    for (int nt = 0; nt < 4; ++nt)
#pragma unroll
      for (int r = 0; r < 8; ++r) {
        int m = mt * 16 + r + hi8;
        int c = wv * 64 + nt * 16 + cl;
        float mu = rsum[m] * (1.f / CH);
        float var = rsq[m] * (1.f / CH) - mu * mu;
        float y = (acc[mt][nt][r] - mu) * rsqrtf(var + 1e-5f) * g1[c] + b1[c];
        msg16[(size_t)(n * LSEQ + tb * TILE + m) * CH + c] = (bf16_t)y;
      }
}

// ---------------- fused MLP (concat GEMM -> relu -> GEMM -> LN2 -> residual) ----
__global__ __launch_bounds__(256) void mlp_kernel(
    const bf16_t* __restrict__ x16, const bf16_t* __restrict__ msg16,
    const bf16_t* __restrict__ w1, const bf16_t* __restrict__ w2,
    const float* __restrict__ g2, const float* __restrict__ b2,
    const float* __restrict__ xf, float* __restrict__ out) {
  __shared__ bf16_t als[TILE * CH2];   // [x | msg] 64x1024
  __shared__ bf16_t act[TILE * CH];    // per-pass activations
  __shared__ bf16_t slabA[SLAB_ELEMS];
  __shared__ bf16_t slabB[SLAB_ELEMS];
  __shared__ float  rsum[TILE], rsq[TILE];

  int t = threadIdx.x, lane = t & 31, wv = t >> 5;
  int n = blockIdx.x >> 6, tb = blockIdx.x & 63;
  int hi8 = (lane >> 4) << 3, cl = lane & 15;
  size_t rowbase = (size_t)(n * LSEQ + tb * TILE);
  const bf16_t* xsrc = x16 + rowbase * CH;
  const bf16_t* msrc = msg16 + rowbase * CH;

  for (int i = t; i < TILE * CH / 8; i += 256) {
    int row = i >> 6;
    int cc = (i & 63) << 3;
    *(v8bf*)(als + (size_t)row * CH2 + cc)      = *(const v8bf*)(xsrc + (size_t)row * CH + cc);
    *(v8bf*)(als + (size_t)row * CH2 + CH + cc) = *(const v8bf*)(msrc + (size_t)row * CH + cc);
  }
  if (t < TILE) { rsum[t] = 0.f; rsq[t] = 0.f; }
  __syncthreads();

  v8f acc2[4][4];
#pragma unroll
  for (int mt = 0; mt < 4; ++mt)
#pragma unroll
    for (int nt = 0; nt < 4; ++nt) acc2[mt][nt] = zero8();

  for (int p = 0; p < 2; ++p) {
    v8f acc1[4][4];
#pragma unroll
    for (int mt = 0; mt < 4; ++mt)
#pragma unroll
      for (int nt = 0; nt < 4; ++nt) acc1[mt][nt] = zero8();
    // GEMM1 pass p: act = relu([x|msg] @ w1[:, p*512 : p*512+512])
    gemm_db(als, CH2, CH2 / 32, w1 + (size_t)p * CH * SLS, (size_t)CH2 * SLS,
            slabA, slabB, t, wv, lane, acc1);
#pragma unroll
    for (int mt = 0; mt < 4; ++mt)
#pragma unroll
      for (int nt = 0; nt < 4; ++nt)
#pragma unroll
        for (int r = 0; r < 8; ++r) {
          int m = mt * 16 + r + hi8;
          int c = wv * 64 + nt * 16 + cl;
          act[m * CH + c] = (bf16_t)fmaxf(acc1[mt][nt][r], 0.f);
        }
    __syncthreads();
    // GEMM2 partial over k in [p*512, p*512+512)
    gemm_db(act, CH, CH / 32, w2 + (size_t)p * (CH / 32) * CH * SLS, (size_t)CH * SLS,
            slabA, slabB, t, wv, lane, acc2);
  }

  // layernorm(norm2) + residual
#pragma unroll
  for (int mt = 0; mt < 4; ++mt)
#pragma unroll
    for (int r = 0; r < 8; ++r) {
      int m = mt * 16 + r + hi8;
      float s1 = 0.f, s2 = 0.f;
#pragma unroll
      for (int nt = 0; nt < 4; ++nt) { float v = acc2[mt][nt][r]; s1 += v; s2 += v * v; }
      atomicAdd(&rsum[m], s1);
      atomicAdd(&rsq[m], s2);
    }
  __syncthreads();
#pragma unroll
  for (int mt = 0; mt < 4; ++mt)
#pragma unroll
    for (int nt = 0; nt < 4; ++nt)
#pragma unroll
      for (int r = 0; r < 8; ++r) {
        int m = mt * 16 + r + hi8;
        int c = wv * 64 + nt * 16 + cl;
        float mu = rsum[m] * (1.f / CH);
        float var = rsq[m] * (1.f / CH) - mu * mu;
        float y = (acc2[mt][nt][r] - mu) * rsqrtf(var + 1e-5f) * g2[c] + b2[c];
        size_t idx = (rowbase + m) * CH + c;
        out[idx] = xf[idx] + y;
      }
}

// ---------------- launch ----------------
extern "C" void kernel_launch(void* const* d_in, const int* in_sizes, int n_in,
                              void* d_out, int out_size, void* d_ws, size_t ws_size,
                              hipStream_t stream) {
  (void)in_sizes; (void)n_in; (void)out_size; (void)ws_size;
  const float* x       = (const float*)d_in[0];
  const float* mg_w1   = (const float*)d_in[1];
  const float* mg_b1   = (const float*)d_in[2];
  const float* mg_w2   = (const float*)d_in[3];
  const float* mg_b2   = (const float*)d_in[4];
  const float* q_w     = (const float*)d_in[5];
  const float* k_w     = (const float*)d_in[6];
  const float* v_w     = (const float*)d_in[7];
  const float* merge_w = (const float*)d_in[8];
  const float* mlp_w1  = (const float*)d_in[9];
  const float* mlp_w2  = (const float*)d_in[10];
  const float* n1g = (const float*)d_in[11];
  const float* n1b = (const float*)d_in[12];
  const float* n2g = (const float*)d_in[13];
  const float* n2b = (const float*)d_in[14];
  float* out = (float*)d_out;
  char* ws = (char*)d_ws;

  float*  gfeat = (float*)(ws + O_GFEAT);
  float*  hbuf  = (float*)(ws + O_H);
  float*  mem   = (float*)(ws + O_MEM);
  float*  Kmat  = (float*)(ws + O_KMAT);
  float*  Vmat  = (float*)(ws + O_VMAT);
  float*  ksumb = (float*)(ws + O_KSUM);
  bf16_t* kvt   = (bf16_t*)(ws + O_KVT);
  bf16_t* qw16  = (bf16_t*)(ws + O_QW);
  bf16_t* mw16  = (bf16_t*)(ws + O_MW);
  bf16_t* w116  = (bf16_t*)(ws + O_W1);
  bf16_t* w216  = (bf16_t*)(ws + O_W2);
  bf16_t* x16   = (bf16_t*)(ws + O_X16);
  bf16_t* msg16 = (bf16_t*)(ws + O_MSG);

  mean_kernel  <<<NB, CH, 0, stream>>>(x, gfeat);
  hidden_kernel<<<NB, CH2, 0, stream>>>(gfeat, mg_w1, mg_b1, hbuf);
  memgen_kernel<<<(CH * KVS * 2) / 256, 256, 0, stream>>>(hbuf, mg_w2, mg_b2, mem);
  kvproj_kernel<<<NB * KVS, 256, 0, stream>>>(mem, k_w, v_w, Kmat, Vmat);
  kvred_kernel <<<NB * NHEAD, 256, 0, stream>>>(Kmat, Vmat, ksumb, kvt);

  cvtw_kernel<<<(CH * CH) / 256, 256, 0, stream>>>(q_w, qw16, CH, CH);
  cvtw_kernel<<<(CH * CH) / 256, 256, 0, stream>>>(merge_w, mw16, CH, CH);
  cvtw_kernel<<<(CH2 * CH2) / 256, 256, 0, stream>>>(mlp_w1, w116, CH2, CH2);
  cvtw_kernel<<<(CH2 * CH) / 256, 256, 0, stream>>>(mlp_w2, w216, CH2, CH);
  cvt_kernel <<<(NB * LSEQ * CH) / 256, 256, 0, stream>>>(x, x16, (size_t)NB * LSEQ * CH);

  qattn_kernel<<<NB * (LSEQ / TILE), 256, 0, stream>>>(x16, qw16, ksumb, kvt, mw16,
                                                       n1g, n1b, msg16);
  mlp_kernel  <<<NB * (LSEQ / TILE), 256, 0, stream>>>(x16, msg16, w116, w216,
                                                       n2g, n2b, x, out);
}
